// ContrastiveLoss_26757646254505
// MI455X (gfx1250) — matrix operand order repaired and checked by reading, not compiled, not run on hardware
//
#include <hip/hip_runtime.h>
#include <hip/hip_bf16.h>

// CDNA5 / gfx1250 wave32 WMMA implementation of the masked contrastive loss.
typedef __bf16 bf16_t;
typedef __attribute__((ext_vector_type(16))) __bf16 v16bf;
typedef __attribute__((ext_vector_type(8)))  __bf16 v8bf;
typedef __attribute__((ext_vector_type(8)))  float  v8f;

#define BATCH 4096
#define DIM   768
#define CL_EPS 1e-6f
#define MARGIN 1.0f

// ---------------------------------------------------------------------------
// Kernel 1: split fp32 embeddings into bf16 hi/lo pair (hi + lo ~= fp32).
// ---------------------------------------------------------------------------
__global__ __launch_bounds__(256) void split_convert(const float* __restrict__ emb,
                                                     bf16_t* __restrict__ hi,
                                                     bf16_t* __restrict__ lo) {
  for (int i = blockIdx.x * 256 + threadIdx.x; i < BATCH * DIM; i += gridDim.x * 256) {
    float x = emb[i];
    bf16_t h = (bf16_t)x;
    hi[i] = h;
    lo[i] = (bf16_t)(x - (float)h);
  }
}

// ---------------------------------------------------------------------------
// Kernel 2: per-row sum of squares, plain sum (for the eps correction), masks.
// ---------------------------------------------------------------------------
__global__ __launch_bounds__(256) void row_stats(const float* __restrict__ emb,
                                                 float* __restrict__ s2,
                                                 float* __restrict__ rowSum,
                                                 const int* __restrict__ labels,
                                                 const int* __restrict__ depIdx,
                                                 const int* __restrict__ suiIdx,
                                                 float* __restrict__ depMask,
                                                 float* __restrict__ suiMask) {
  const int row = blockIdx.x;
  const float* p = emb + (size_t)row * DIM;
  float ss = 0.0f, sm = 0.0f;
  for (int i = threadIdx.x; i < DIM; i += 256) {
    float x = p[i];
    ss += x * x;
    sm += x;
  }
  __shared__ float rs[256], rm[256];
  rs[threadIdx.x] = ss;
  rm[threadIdx.x] = sm;
  __syncthreads();
  for (int s = 128; s > 0; s >>= 1) {
    if (threadIdx.x < s) {
      rs[threadIdx.x] += rs[threadIdx.x + s];
      rm[threadIdx.x] += rm[threadIdx.x + s];
    }
    __syncthreads();
  }
  if (threadIdx.x == 0) {
    s2[row]     = rs[0];
    rowSum[row] = rm[0];
    const int lab = labels[row];
    depMask[row] = (lab == *depIdx) ? 1.0f : 0.0f;
    suiMask[row] = (lab == *suiIdx) ? 1.0f : 0.0f;
  }
}

// ---------------------------------------------------------------------------
// Kernel 3: per-16-row tile "any member of class" flags (enables tile skip).
// ---------------------------------------------------------------------------
__global__ __launch_bounds__(256) void tile_any(const float* __restrict__ depMask,
                                                const float* __restrict__ suiMask,
                                                int* __restrict__ depAny,
                                                int* __restrict__ suiAny) {
  const int t = threadIdx.x;  // 0..255 -> 256 tiles of 16 rows
  int d = 0, s = 0;
  for (int i = 0; i < 16; ++i) {
    d |= (depMask[t * 16 + i] != 0.0f);
    s |= (suiMask[t * 16 + i] != 0.0f);
  }
  depAny[t] = d;
  suiAny[t] = s;
}

// ---------------------------------------------------------------------------
// Kernel 4: fused WMMA Gram-matrix + hinge + mask reduction.
// Block = 256 threads = 8 waves; block tile = 64(M) x 32(N); wave tile 16x16.
// Split-bf16: dot = aHi*bHi + aHi*bLo + aLo*bHi  (~fp32 precision).
// ---------------------------------------------------------------------------
__global__ __launch_bounds__(256) void gemm_hinge(
    const bf16_t* __restrict__ eHi, const bf16_t* __restrict__ eLo,
    const float* __restrict__ s2, const float* __restrict__ rowSum,
    const float* __restrict__ depMask, const float* __restrict__ suiMask,
    const int* __restrict__ depAny, const int* __restrict__ suiAny,
    float* __restrict__ partials) {
  const int lane  = threadIdx.x & 31;
  const int wave  = threadIdx.x >> 5;
  const int tileM = blockIdx.x * 4 + (wave & 3);   // 16-row tile index (0..255)
  const int tileN = blockIdx.y * 2 + (wave >> 2);  // 16-col tile index (0..255)

  float part = 0.0f;
  if (depAny[tileM] && suiAny[tileN]) {
    const int laneLo = lane & 15;
    const int laneHi = lane >> 4;
    // A-matrix (16x32 bf16): lanes 0-15 carry K {0..7,16..23}, lanes 16-31 {8..15,24..31}
    const size_t aBase = (size_t)(tileM * 16 + laneLo) * DIM + laneHi * 8;
    // B-matrix (32x16): lane carries column n = embedding row n, K {0..7,16..23} / {8..15,24..31}
    const size_t bBase = (size_t)(tileN * 16 + laneLo) * DIM + laneHi * 8;

    v8f acc = {};
    for (int k = 0; k < DIM; k += 32) {
      union U16 { v16bf v; v8bf h[2]; };
      U16 aH, aL, bH, bL;
      aH.h[0] = *(const v8bf*)(eHi + aBase + k);
      aH.h[1] = *(const v8bf*)(eHi + aBase + k + 16);
      aL.h[0] = *(const v8bf*)(eLo + aBase + k);
      aL.h[1] = *(const v8bf*)(eLo + aBase + k + 16);
      bH.h[0] = *(const v8bf*)(eHi + bBase + k);
      bH.h[1] = *(const v8bf*)(eHi + bBase + k + 16);
      bL.h[0] = *(const v8bf*)(eLo + bBase + k);
      bL.h[1] = *(const v8bf*)(eLo + bBase + k + 16);
      acc = __builtin_amdgcn_wmma_f32_16x16x32_bf16(false, aL.v, false, bH.v,
                                                    (short)0, acc, false, false);
      acc = __builtin_amdgcn_wmma_f32_16x16x32_bf16(false, aH.v, false, bL.v,
                                                    (short)0, acc, false, false);
      acc = __builtin_amdgcn_wmma_f32_16x16x32_bf16(false, aH.v, false, bH.v,
                                                    (short)0, acc, false, false);
    }

    // Epilogue: C/D layout -> lane holds column n = laneLo, rows m = v + 8*laneHi.
    const int   n   = tileN * 16 + laneLo;
    const float sm  = suiMask[n];
    if (sm != 0.0f) {
      const float s2n = s2[n];
      const float rsn = rowSum[n];
      const float cD  = (float)DIM * CL_EPS * CL_EPS;
#pragma unroll
      for (int v = 0; v < 8; ++v) {
        const int m = tileM * 16 + v + laneHi * 8;
        // ||e_m + eps||^2 + ||e_n||^2 - 2 (e_m + eps) . e_n
        float dist2 = s2[m] + s2n + 2.0f * CL_EPS * rowSum[m] + cD
                    - 2.0f * acc[v] - 2.0f * CL_EPS * rsn;
        dist2 = fmaxf(dist2, 1e-12f);
        const float hinge = fmaxf(MARGIN - sqrtf(dist2), 0.0f);
        part += hinge * depMask[m];
      }
      part *= sm;
    }
  }

  // Deterministic block reduction of hinge contributions.
  __shared__ float red[256];
  red[threadIdx.x] = part;
  __syncthreads();
  for (int s = 128; s > 0; s >>= 1) {
    if (threadIdx.x < s) red[threadIdx.x] += red[threadIdx.x + s];
    __syncthreads();
  }
  if (threadIdx.x == 0) partials[blockIdx.y * gridDim.x + blockIdx.x] = red[0];
}

// ---------------------------------------------------------------------------
// Kernel 5: final deterministic reduce; count = nDep * nSui.
// ---------------------------------------------------------------------------
__global__ __launch_bounds__(256) void finalize(const float* __restrict__ partials,
                                                int nPart,
                                                const int* __restrict__ labels,
                                                const int* __restrict__ depIdx,
                                                const int* __restrict__ suiIdx,
                                                float* __restrict__ out) {
  float tot = 0.0f;
  int nd = 0, ns = 0;
  const int di = *depIdx, si = *suiIdx;
  for (int i = threadIdx.x; i < nPart; i += 256) tot += partials[i];
  for (int i = threadIdx.x; i < BATCH; i += 256) {
    const int l = labels[i];
    nd += (l == di);
    ns += (l == si);
  }
  __shared__ float rt[256];
  __shared__ int   rd[256], rn[256];
  rt[threadIdx.x] = tot;
  rd[threadIdx.x] = nd;
  rn[threadIdx.x] = ns;
  __syncthreads();
  for (int s = 128; s > 0; s >>= 1) {
    if (threadIdx.x < s) {
      rt[threadIdx.x] += rt[threadIdx.x + s];
      rd[threadIdx.x] += rd[threadIdx.x + s];
      rn[threadIdx.x] += rn[threadIdx.x + s];
    }
    __syncthreads();
  }
  if (threadIdx.x == 0) {
    const float count = (float)rd[0] * (float)rn[0];
    out[0] = (count > 0.0f) ? rt[0] / count : 0.0f;
  }
}

// ---------------------------------------------------------------------------
extern "C" void kernel_launch(void* const* d_in, const int* in_sizes, int n_in,
                              void* d_out, int out_size, void* d_ws, size_t ws_size,
                              hipStream_t stream) {
  (void)in_sizes; (void)n_in; (void)out_size; (void)ws_size;
  const float* emb    = (const float*)d_in[0];
  const int*   labels = (const int*)d_in[1];
  const int*   depIdx = (const int*)d_in[2];
  const int*   suiIdx = (const int*)d_in[3];

  char* ws = (char*)d_ws;
  const size_t embBytes = (size_t)BATCH * DIM * sizeof(bf16_t);  // 6,291,456
  bf16_t* eHi    = (bf16_t*)(ws);
  bf16_t* eLo    = (bf16_t*)(ws + embBytes);
  float*  s2     = (float*)(ws + 2 * embBytes);
  float*  rowSum = s2 + BATCH;
  float*  depM   = rowSum + BATCH;
  float*  suiM   = depM + BATCH;
  int*    depAny = (int*)(suiM + BATCH);
  int*    suiAny = depAny + 256;
  float*  parts  = (float*)(suiAny + 256);

  split_convert<<<1536, 256, 0, stream>>>(emb, eHi, eLo);
  row_stats<<<BATCH, 256, 0, stream>>>(emb, s2, rowSum, labels, depIdx, suiIdx,
                                       depM, suiM);
  tile_any<<<1, 256, 0, stream>>>(depM, suiM, depAny, suiAny);

  dim3 grid(BATCH / 64, BATCH / 32);  // 64 x 128 blocks, 8 waves each
  gemm_hinge<<<grid, 256, 0, stream>>>(eHi, eLo, s2, rowSum, depM, suiM,
                                       depAny, suiAny, parts);
  finalize<<<1, 256, 0, stream>>>(parts, (BATCH / 64) * (BATCH / 32), labels,
                                  depIdx, suiIdx, (float*)d_out);
}
